// DeepSeekHybridAttention_34840774705580
// MI455X (gfx1250) — matrix operand order repaired and checked by reading, not compiled, not run on hardware
//
#include <hip/hip_runtime.h>

// ---------------- problem constants ----------------
constexpr int Bn   = 2;
constexpr int Tn   = 2048;
constexpr int Cn   = 2048;
constexpr int NHn  = 16;
constexpr int NKVn = 8;
constexpr int HDn  = 128;
constexpr int Gn   = 512;      // T / RATIO
constexpr int RATn = 4;
constexpr int IDXNH = 16;
constexpr int IDXHD = 64;
constexpr int TOPKn = 64;
constexpr int MT   = Bn * Tn;  // 4096
constexpr float NEGF = -3.402823466e38f;

// GEMM tiling
constexpr int KC  = 128;        // K-chunk staged in LDS
constexpr int LDB = KC + 8;     // padded LDS stride (bank-conflict-free, 16B aligned)

// ---------------- WMMA types ----------------
typedef __attribute__((ext_vector_type(16))) __bf16 v16bf;
typedef __attribute__((ext_vector_type(8)))  float  v8f;

union FragBF16 {
    v16bf v;
    unsigned short u[16];
    uint4 q[2];
};

__device__ __forceinline__ unsigned short f2bf_rne(float f) {
    unsigned u = __float_as_uint(f);
    unsigned r = (u + 0x7FFFu + ((u >> 16) & 1u)) >> 16;
    return (unsigned short)r;
}

// ---------------- f32 -> bf16 (flat) ----------------
__global__ void k_f32_to_bf16(const float* __restrict__ in,
                              unsigned short* __restrict__ outp, size_t n) {
    size_t i = (size_t)blockIdx.x * blockDim.x + threadIdx.x;
    if (i < n) outp[i] = f2bf_rne(in[i]);
}

// ---------------- f32 (R x C) -> bf16 transposed (C x R) ----------------
__global__ void k_f32_to_bf16_T(const float* __restrict__ in,
                                unsigned short* __restrict__ outp,
                                int R, int Ccols) {
    __shared__ unsigned short tile[32][33];
    int c0 = blockIdx.x * 32, r0 = blockIdx.y * 32;
    for (int dy = 0; dy < 32; dy += 8) {
        int r = r0 + threadIdx.y + dy, c = c0 + threadIdx.x;
        tile[threadIdx.y + dy][threadIdx.x] = f2bf_rne(in[(size_t)r * Ccols + c]);
    }
    __syncthreads();
    for (int dy = 0; dy < 32; dy += 8) {
        int c = c0 + threadIdx.y + dy, r = r0 + threadIdx.x;
        outp[(size_t)c * R + r] = tile[threadIdx.x][threadIdx.y + dy];
    }
}

// ---------------- bf16 NT GEMM: C(M,N) = A(M,K) * Bt(N,K)^T ----------------
// 256 threads = 8 waves; block tile 128(M) x 64(N); wave tile 16 x 64 (4 accs).
// B panel staged to LDS via global_load_async_to_lds_b128 (ASYNCcnt), double
// buffered in K-chunks of 128; fragments consumed with ds_load_b128.
__global__ __launch_bounds__(256) void k_gemm_nt(
    const unsigned short* __restrict__ A, const unsigned short* __restrict__ Bt,
    float* __restrict__ C, int M, int N, int K,
    long long sA, long long sB, long long sC) {
    __shared__ alignas(16) unsigned short bsm[2][64 * LDB];

    const unsigned short* Ab = A + sA * blockIdx.z;
    const unsigned short* Bb = Bt + sB * blockIdx.z;
    float* Cb = C + sC * blockIdx.z;

    const int tid  = threadIdx.x;
    const int wave = tid >> 5;
    const int lane = tid & 31;
    const int lm   = lane & 15;
    const int half = lane >> 4;
    const int m0   = blockIdx.y * 128 + wave * 16;
    const int n0   = blockIdx.x * 64;

    // Per-thread async staging plan: 4 x b128 per 64x128 chunk.
    // li = i*256 + tid ; row = li/16 (16 x b128 per row), col = (li%16)*8.
    unsigned ldsoff[2][4];
    unsigned long long gsrc[4];
    #pragma unroll
    for (int i = 0; i < 4; ++i) {
        int li  = i * 256 + tid;
        int row = li >> 4;
        int col = (li & 15) * 8;
        gsrc[i] = (unsigned long long)(uintptr_t)(Bb + (size_t)(n0 + row) * K + col);
        ldsoff[0][i] = (unsigned)(uintptr_t)(&bsm[0][row * LDB + col]);
        ldsoff[1][i] = (unsigned)(uintptr_t)(&bsm[1][row * LDB + col]);
    }

    v8f acc[4];
    v8f zero = {};
    #pragma unroll
    for (int j = 0; j < 4; ++j) acc[j] = zero;

    const int NC = K / KC;

    auto issue = [&](int c) {
        int bi = c & 1;
        #pragma unroll
        for (int i = 0; i < 4; ++i) {
            unsigned long long gp = gsrc[i] + (unsigned long long)c * (KC * 2);
            asm volatile("global_load_async_to_lds_b128 %0, %1, off"
                         :: "v"(ldsoff[bi][i]), "v"(gp)
                         : "memory");
        }
    };

    issue(0);
    asm volatile("s_wait_asynccnt 0x0" ::: "memory");
    __syncthreads();

    for (int c = 0; c < NC; ++c) {
        if (c + 1 < NC) issue(c + 1);

        const int bi = c & 1;
        const unsigned short* arow = Ab + (size_t)(m0 + lm) * K + c * KC;
        __builtin_prefetch(arow + KC, 0, 1);  // next A chunk
        #pragma unroll
        for (int k0 = 0; k0 < KC; k0 += 32) {
            FragBF16 a;
            a.q[0] = *(const uint4*)(arow + k0 + half * 8);
            a.q[1] = *(const uint4*)(arow + k0 + 16 + half * 8);
            FragBF16 b[4];
            #pragma unroll
            for (int j = 0; j < 4; ++j) {
                const uint4* bp =
                    (const uint4*)&bsm[bi][(j * 16 + lm) * LDB + k0 + half * 8];
                b[j].q[0] = bp[0];
                b[j].q[1] = bp[2];
            }
            #pragma unroll
            for (int j = 0; j < 4; ++j)
                acc[j] = __builtin_amdgcn_wmma_f32_16x16x32_bf16(
                    false, a.v, false, b[j].v, (short)0, acc[j], false, false);
        }

        if (c + 1 < NC) {
            asm volatile("s_wait_asynccnt 0x0" ::: "memory");
            __syncthreads();
        }
    }

    #pragma unroll
    for (int i = 0; i < 8; ++i) {
        int row = m0 + half * 8 + i;
        float* crow = Cb + (size_t)row * N + n0 + lm;
        crow[0]  = acc[0][i];
        crow[16] = acc[1][i];
        crow[32] = acc[2][i];
        crow[48] = acc[3][i];
    }
}

// ---------------- partial RoPE (last 64 dims of each 128-dim head) ----------
__global__ void k_rope(const float* __restrict__ src, float* __restrict__ dst,
                       const float* __restrict__ cosT, const float* __restrict__ sinT,
                       int rows, int heads, int posMod, int posMul, int posAdd) {
    int idx = blockIdx.x * blockDim.x + threadIdx.x;
    int total = rows * heads * 96;
    if (idx >= total) return;
    int j  = idx % 96;
    int rh = idx / 96;
    int h  = rh % heads;
    int r  = rh / heads;
    int pos = (r % posMod) * posMul + posAdd;
    size_t base = ((size_t)r * heads + h) * HDn;
    if (j < 64) {
        dst[base + j] = src[base + j];
    } else {
        int i = j - 64;  // 0..31
        float c = cosT[pos * 64 + i], s = sinT[pos * 64 + i];
        float x1 = src[base + 64 + i], x2 = src[base + 96 + i];
        dst[base + 64 + i] =  x1 * c + x2 * s;
        dst[base + 96 + i] = -x1 * s + x2 * c;
    }
}

// ---------------- compress gate: softmax over RATIO, weighted sum ----------
__global__ void k_cgate(const float* __restrict__ kb, const float* __restrict__ vb,
                        const float* __restrict__ gb, const float* __restrict__ ape,
                        float* __restrict__ ck, float* __restrict__ cv) {
    int idx = blockIdx.x * blockDim.x + threadIdx.x;
    if (idx >= Bn * Gn * NKVn * HDn) return;
    int d  = idx & (HDn - 1);
    int kv = (idx >> 7) & (NKVn - 1);
    int bg = idx >> 10;
    int g  = bg % Gn;
    int b  = bg / Gn;
    size_t col = (size_t)kv * HDn + d;
    float gv[RATn], m = NEGF;
    #pragma unroll
    for (int r = 0; r < RATn; ++r) {
        size_t row = (size_t)(b * Tn + g * RATn + r);
        gv[r] = gb[row * (NKVn * HDn) + col] + ape[((size_t)r * NKVn + kv) * HDn + d];
        m = fmaxf(m, gv[r]);
    }
    float den = 0.f;
    #pragma unroll
    for (int r = 0; r < RATn; ++r) { gv[r] = __expf(gv[r] - m); den += gv[r]; }
    float sk = 0.f, sv = 0.f;
    #pragma unroll
    for (int r = 0; r < RATn; ++r) {
        size_t row = (size_t)(b * Tn + g * RATn + r);
        float w = gv[r] / den;
        sk += w * kb[row * (NKVn * HDn) + col];
        sv += w * vb[row * (NKVn * HDn) + col];
    }
    ck[(size_t)bg * (NKVn * HDn) + col] = sk;
    cv[(size_t)bg * (NKVn * HDn) + col] = sv;
}

// ---------------- indexer gate + rms_norm (per (b,g,h), 64 lanes) ----------
__global__ void k_igate(const float* __restrict__ ikb, const float* __restrict__ igb,
                        const float* __restrict__ iape, float* __restrict__ ikeys) {
    int h  = blockIdx.x & (IDXNH - 1);
    int bg = blockIdx.x >> 4;
    int g  = bg % Gn;
    int b  = bg / Gn;
    int d  = threadIdx.x;
    size_t col = (size_t)h * IDXHD + d;
    float gv[RATn], m = NEGF;
    #pragma unroll
    for (int r = 0; r < RATn; ++r) {
        size_t row = (size_t)(b * Tn + g * RATn + r);
        gv[r] = igb[row * (IDXNH * IDXHD) + col] + iape[((size_t)r * IDXNH + h) * IDXHD + d];
        m = fmaxf(m, gv[r]);
    }
    float den = 0.f;
    #pragma unroll
    for (int r = 0; r < RATn; ++r) { gv[r] = __expf(gv[r] - m); den += gv[r]; }
    float s = 0.f;
    #pragma unroll
    for (int r = 0; r < RATn; ++r) {
        size_t row = (size_t)(b * Tn + g * RATn + r);
        s += (gv[r] / den) * ikb[row * (IDXNH * IDXHD) + col];
    }
    __shared__ float red[64];
    red[d] = s * s;
    __syncthreads();
    for (int off = 32; off > 0; off >>= 1) {
        if (d < off) red[d] += red[d + off];
        __syncthreads();
    }
    float inv = rsqrtf(red[0] * (1.0f / IDXHD) + 1e-6f);
    ikeys[(size_t)bg * (IDXNH * IDXHD) + col] = s * inv;
}

// ---------------- in-place rms_norm on rows of given width ----------------
__global__ void k_rms_rows(float* __restrict__ x, int width) {
    __shared__ float red[256];
    size_t base = (size_t)blockIdx.x * width;
    float ss = 0.f;
    for (int i = threadIdx.x; i < width; i += blockDim.x) {
        float v = x[base + i];
        ss += v * v;
    }
    red[threadIdx.x] = ss;
    __syncthreads();
    for (int off = blockDim.x >> 1; off > 0; off >>= 1) {
        if ((int)threadIdx.x < off) red[threadIdx.x] += red[threadIdx.x + off];
        __syncthreads();
    }
    float inv = rsqrtf(red[0] / (float)width + 1e-6f);
    for (int i = threadIdx.x; i < width; i += blockDim.x) x[base + i] *= inv;
}

// ---------------- top-64 per (b,t) over causal groups -> 512-bit mask -------
__global__ __launch_bounds__(128) void k_topk(const float* __restrict__ iscore,
                                              unsigned* __restrict__ maskb) {
    int bt = blockIdx.x;
    int t  = bt % Tn;
    __shared__ float sc[Gn];
    __shared__ float rmax[128];
    __shared__ int   ridx[128];
    __shared__ unsigned bits[16];
    int gmax = (t >= 3) ? ((t - 3) >> 2) : -1;  // group_end 4g+3 <= t
    for (int g = threadIdx.x; g < Gn; g += 128)
        sc[g] = (g <= gmax) ? iscore[(size_t)bt * Gn + g] : NEGF;
    if (threadIdx.x < 16) bits[threadIdx.x] = 0u;
    __syncthreads();
    for (int it = 0; it < TOPKn; ++it) {
        float lm = NEGF; int li = -1;
        for (int g = threadIdx.x; g < Gn; g += 128)
            if (sc[g] > lm) { lm = sc[g]; li = g; }
        rmax[threadIdx.x] = lm; ridx[threadIdx.x] = li;
        __syncthreads();
        for (int off = 64; off > 0; off >>= 1) {
            if ((int)threadIdx.x < off && rmax[threadIdx.x + off] > rmax[threadIdx.x]) {
                rmax[threadIdx.x] = rmax[threadIdx.x + off];
                ridx[threadIdx.x] = ridx[threadIdx.x + off];
            }
            __syncthreads();
        }
        float mv = rmax[0]; int selg = ridx[0];
        if (mv == NEGF) break;
        if (threadIdx.x == 0) {
            bits[selg >> 5] |= 1u << (selg & 31);
            sc[selg] = NEGF;
        }
        __syncthreads();
    }
    if (threadIdx.x < 16) maskb[(size_t)bt * 16 + threadIdx.x] = bits[threadIdx.x];
}

// ---------------- gathered sparse attention with sink ----------------------
__global__ __launch_bounds__(128) void k_attn(const float* __restrict__ q,
                                              const float* __restrict__ ckr,
                                              const float* __restrict__ cv,
                                              const unsigned* __restrict__ maskb,
                                              const float* __restrict__ sink,
                                              float* __restrict__ outp) {
    int bt = blockIdx.x;
    int h  = blockIdx.y;
    int b  = bt / Tn;
    int tid = threadIdx.x;
    __shared__ float qv[HDn];
    __shared__ short sel[TOPKn];
    __shared__ float sc[TOPKn];
    __shared__ int   ncnt;
    __shared__ float pden;

    qv[tid] = q[(size_t)bt * Cn + (size_t)h * HDn + tid];
    if (tid == 0) {
        int c = 0;
        for (int w = 0; w < 16; ++w) {
            unsigned bw = maskb[(size_t)bt * 16 + w];
            while (bw) {
                int bpos = __builtin_ctz(bw);
                bw &= bw - 1;
                sel[c++] = (short)(w * 32 + bpos);
            }
        }
        ncnt = c;
    }
    __syncthreads();
    int n = ncnt;
    int kv = h >> 1;  // rep = NH/NKV = 2
    for (int i = tid; i < n; i += 128) {
        const float* krow = ckr + ((size_t)(b * Gn + sel[i])) * (NKVn * HDn) + (size_t)kv * HDn;
        float acc = 0.f;
        #pragma unroll 8
        for (int d = 0; d < HDn; ++d) acc += qv[d] * krow[d];
        sc[i] = acc * 0.08838834764831845f;  // HD^-0.5
    }
    __syncthreads();
    if (tid == 0) {
        float m = sink[h];
        for (int i = 0; i < n; ++i) m = fmaxf(m, sc[i]);
        float den = __expf(sink[h] - m);
        for (int i = 0; i < n; ++i) { sc[i] = __expf(sc[i] - m); den += sc[i]; }
        pden = den;
    }
    __syncthreads();
    float acc = 0.f;
    for (int i = 0; i < n; ++i) {
        float p = sc[i] / pden;
        acc += p * cv[((size_t)(b * Gn + sel[i])) * (NKVn * HDn) + (size_t)kv * HDn + tid];
    }
    outp[(size_t)bt * Cn + (size_t)h * HDn + tid] = acc;
}

// ======================== host side ========================
extern "C" void kernel_launch(void* const* d_in, const int* in_sizes, int n_in,
                              void* d_out, int out_size, void* d_ws, size_t ws_size,
                              hipStream_t stream) {
    (void)in_sizes; (void)n_in; (void)out_size; (void)ws_size;
    const float* x     = (const float*)d_in[0];
    const float* cosT  = (const float*)d_in[1];
    const float* sinT  = (const float*)d_in[2];
    const float* q_a_w = (const float*)d_in[3];
    const float* q_b_w = (const float*)d_in[4];
    const float* ck_w  = (const float*)d_in[5];
    const float* cv_w  = (const float*)d_in[6];
    const float* cg_w  = (const float*)d_in[7];
    const float* c_ape = (const float*)d_in[8];
    const float* iq_w  = (const float*)d_in[9];
    const float* ik_w  = (const float*)d_in[10];
    const float* ig_w  = (const float*)d_in[11];
    const float* i_ape = (const float*)d_in[12];
    const float* sink  = (const float*)d_in[13];
    const float* o_wa  = (const float*)d_in[14];
    const float* o_pb  = (const float*)d_in[15];
    float* outp = (float*)d_out;

    char* ws = (char*)d_ws;
    size_t off = 0;
    auto alloc = [&](size_t bytes) -> void* {
        void* p = ws + off;
        off = (off + bytes + 255) & ~(size_t)255;
        return p;
    };

    unsigned short* xb     = (unsigned short*)alloc((size_t)MT * Cn * 2);        // 16 MB
    unsigned short* wslot  = (unsigned short*)alloc((size_t)2048 * 1024 * 2);    // 4 MB (reused)
    float* kf    = (float*)alloc((size_t)MT * 1024 * 4);   // k  -> later ik
    float* vf    = (float*)alloc((size_t)MT * 1024 * 4);   // v  -> later ig
    float* gf    = (float*)alloc((size_t)MT * 1024 * 4);   // g  -> later q_lat
    float* ckf   = (float*)alloc((size_t)Bn * Gn * 1024 * 4);
    float* cvf   = (float*)alloc((size_t)Bn * Gn * 1024 * 4);
    float* ckrf  = (float*)alloc((size_t)Bn * Gn * 1024 * 4);
    float* ikeysf = (float*)alloc((size_t)Bn * Gn * 1024 * 4);
    unsigned short* ikeysb = (unsigned short*)alloc((size_t)Bn * Gn * 1024 * 2);
    float* iqf   = (float*)alloc((size_t)MT * 1024 * 4);
    unsigned short* iqb   = (unsigned short*)alloc((size_t)MT * 1024 * 2);
    float* iscore = (float*)alloc((size_t)Bn * Tn * Gn * 4);
    unsigned* maskb = (unsigned*)alloc((size_t)Bn * Tn * 16 * 4);
    unsigned short* qlatb = (unsigned short*)alloc((size_t)MT * 1024 * 2);
    float* qf    = (float*)alloc((size_t)MT * Cn * 4);     // 32 MB
    unsigned short* aoutb = (unsigned short*)alloc((size_t)MT * Cn * 2);
    float* hf    = (float*)alloc((size_t)MT * 1024 * 4);
    unsigned short* hb    = (unsigned short*)alloc((size_t)MT * 1024 * 2);

    auto conv = [&](const float* src, unsigned short* dst, size_t n) {
        k_f32_to_bf16<<<(unsigned)((n + 255) / 256), 256, 0, stream>>>(src, dst, n);
    };
    auto convT = [&](const float* src, unsigned short* dst, int R, int Ccols) {
        k_f32_to_bf16_T<<<dim3(Ccols / 32, R / 32), dim3(32, 8), 0, stream>>>(src, dst, R, Ccols);
    };
    auto gemm = [&](const unsigned short* A, const unsigned short* Bt, float* Cc,
                    int M, int N, int K, int batch, long long sA, long long sB, long long sC) {
        k_gemm_nt<<<dim3(N / 64, M / 128, batch), 256, 0, stream>>>(A, Bt, Cc, M, N, K, sA, sB, sC);
    };

    // x -> bf16
    conv(x, xb, (size_t)MT * Cn);

    // compressed k / v / g
    convT(ck_w, wslot, Cn, 1024); gemm(xb, wslot, kf, MT, 1024, Cn, 1, 0, 0, 0);
    convT(cv_w, wslot, Cn, 1024); gemm(xb, wslot, vf, MT, 1024, Cn, 1, 0, 0, 0);
    convT(cg_w, wslot, Cn, 1024); gemm(xb, wslot, gf, MT, 1024, Cn, 1, 0, 0, 0);
    k_cgate<<<(Bn * Gn * NKVn * HDn) / 256, 256, 0, stream>>>(kf, vf, gf, c_ape, ckf, cvf);
    k_rope<<<(Bn * Gn * NKVn * 96 + 255) / 256, 256, 0, stream>>>(
        ckf, ckrf, cosT, sinT, Bn * Gn, NKVn, Gn, 4, 3);

    // indexer keys (reuse kf/vf for ik/ig)
    convT(ik_w, wslot, Cn, 1024); gemm(xb, wslot, kf, MT, 1024, Cn, 1, 0, 0, 0);
    convT(ig_w, wslot, Cn, 1024); gemm(xb, wslot, vf, MT, 1024, Cn, 1, 0, 0, 0);
    k_igate<<<Bn * Gn * IDXNH, 64, 0, stream>>>(kf, vf, i_ape, ikeysf);
    conv(ikeysf, ikeysb, (size_t)Bn * Gn * 1024);

    // indexer queries
    convT(iq_w, wslot, Cn, 1024); gemm(xb, wslot, iqf, MT, 1024, Cn, 1, 0, 0, 0);
    k_rms_rows<<<MT * IDXNH, 64, 0, stream>>>(iqf, IDXHD);
    conv(iqf, iqb, (size_t)MT * 1024);

    // index scores (batched over B) + top-k mask
    gemm(iqb, ikeysb, iscore, Tn, Gn, 1024, Bn,
         (long long)Tn * 1024, (long long)Gn * 1024, (long long)Tn * Gn);
    k_topk<<<Bn * Tn, 128, 0, stream>>>(iscore, maskb);

    // main queries: x @ q_a_w @ q_b_w, partial rope
    convT(q_a_w, wslot, Cn, 1024); gemm(xb, wslot, gf, MT, 1024, Cn, 1, 0, 0, 0);
    conv(gf, qlatb, (size_t)MT * 1024);
    convT(q_b_w, wslot, 1024, Cn); gemm(qlatb, wslot, qf, MT, Cn, 1024, 1, 0, 0, 0);
    k_rope<<<(MT * NHn * 96 + 255) / 256, 256, 0, stream>>>(
        qf, qf, cosT, sinT, MT, NHn, Tn, 1, 0);

    // sparse attention -> d_out (used as scratch here)
    k_attn<<<dim3(Bn * Tn, NHn), 128, 0, stream>>>(qf, ckrf, cvf, maskb, sink, outp);

    // output projections: h = attn_out @ o_wa^T ; d_out = rms(h) @ o_pb
    conv(outp, aoutb, (size_t)MT * Cn);
    conv(o_wa, wslot, (size_t)1024 * Cn);                 // o_wa already (N=1024, K=2048)
    gemm(aoutb, wslot, hf, MT, 1024, Cn, 1, 0, 0, 0);
    k_rms_rows<<<MT, 256, 0, stream>>>(hf, 1024);
    conv(hf, hb, (size_t)MT * 1024);
    convT(o_pb, wslot, 1024, Cn);
    gemm(hb, wslot, outp, MT, Cn, 1024, 1, 0, 0, 0);
}
